// HierarchicalChannelAttention_1279900254216
// MI455X (gfx1250) — matrix operand order repaired and checked by reading, not compiled, not run on hardware
//
#include <hip/hip_runtime.h>

// ---------------------------------------------------------------------------
// Types for CDNA5 WMMA (gfx1250, wave32)
// ---------------------------------------------------------------------------
typedef __attribute__((ext_vector_type(16))) __bf16        bf16x16;
typedef __attribute__((ext_vector_type(8)))  float         f32x8;
typedef __attribute__((ext_vector_type(4)))  unsigned int  u32x4;

union Frag { bf16x16 v; u32x4 q[2]; };

__device__ __forceinline__ unsigned short f2bf(float f) {
  unsigned int u = __float_as_uint(f);
  u += 0x7FFFu + ((u >> 16) & 1u);            // round-to-nearest-even
  return (unsigned short)(u >> 16);
}
__device__ __forceinline__ float bf2f(unsigned short h) {
  return __uint_as_float(((unsigned int)h) << 16);
}
__device__ __forceinline__ f32x8 wmma_bf16(const Frag& a, const Frag& b, f32x8 c) {
  // D = A(16x32 bf16) * B(32x16 bf16) + C(16x16 f32)
  return __builtin_amdgcn_wmma_f32_16x16x32_bf16(
      false, a.v, false, b.v, (short)0, c, false, false);
}
// LDS byte offset of a __shared__ object (low 32 bits of the generic address)
__device__ __forceinline__ unsigned int lds_off(const void* p) {
  return (unsigned int)(size_t)p;
}

// ---------------------------------------------------------------------------
// f32 -> bf16 convert
// ---------------------------------------------------------------------------
__global__ __launch_bounds__(256) void f2bf_kernel(
    const float* __restrict__ in, unsigned short* __restrict__ out, int n) {
  int i = blockIdx.x * 256 + threadIdx.x;
  if (i < n) out[i] = f2bf(in[i]);
}

// ---------------------------------------------------------------------------
// GEMM: Y[m,g] = sum_f X[m,f] * W[g,f] + bias[g]   (M=8192, K=N=256)
// Wave computes a 16(m) x 64(n) tile.  MODE: 0=bf16 flat [m,256],
// 1=bf16 head-split [(b*4+h)*2048+c][64], 2=f32 flat.
// ---------------------------------------------------------------------------
template <int MODE>
__global__ __launch_bounds__(256) void gemm_bias_kernel(
    const unsigned short* __restrict__ X,   // [8192][256] bf16
    const unsigned short* __restrict__ W,   // [256][256]  bf16  (row g, col f)
    const float* __restrict__ bias,         // [256]
    void* __restrict__ Y) {
  const int lane = threadIdx.x & 31;
  const int wave = threadIdx.x >> 5;
  const int wid  = blockIdx.x * 8 + wave;   // 2048 waves
  const int m0   = (wid >> 2) * 16;
  const int n0   = (wid & 3) * 64;
  const int l16  = lane & 15;
  const int h    = lane >> 4;

  f32x8 acc[4] = {};
  const unsigned short* xrow = X + (size_t)(m0 + l16) * 256;
  for (int f0 = 0; f0 < 256; f0 += 32) {
    Frag a;                                  // A: 16x32, lane=row, K chunks
    a.q[0] = *(const u32x4*)(xrow + f0 + h * 16);
    a.q[1] = *(const u32x4*)(xrow + f0 + h * 16 + 8);
#pragma unroll
    for (int nt = 0; nt < 4; ++nt) {         // 4 independent WMMA chains
      const unsigned short* wrow = W + (size_t)(n0 + nt * 16 + l16) * 256 + f0 + h * 16;
      Frag b;                                // B: col n = W row g, contiguous
      b.q[0] = ((const u32x4*)wrow)[0];
      b.q[1] = ((const u32x4*)wrow)[1];
      acc[nt] = wmma_bf16(a, b, acc[nt]);
    }
  }
#pragma unroll
  for (int nt = 0; nt < 4; ++nt) {
    const int g = n0 + nt * 16 + l16;
    const float bv = bias[g];
#pragma unroll
    for (int j = 0; j < 8; ++j) {
      const int m = m0 + j + h * 8;
      const float val = acc[nt][j] + bv;
      if (MODE == 0) {
        ((unsigned short*)Y)[(size_t)m * 256 + g] = f2bf(val);
      } else if (MODE == 1) {
        const int head = g >> 6, d = g & 63;
        const int b_ = m >> 11, c = m & 2047;
        ((unsigned short*)Y)[(((size_t)(b_ * 4 + head)) * 2048 + c) * 64 + d] = f2bf(val);
      } else {
        ((float*)Y)[(size_t)m * 256 + g] = val;
      }
    }
  }
}

// ---------------------------------------------------------------------------
// Local attention with exact top-5 threshold masking.
// One block (8 waves) per (bh, 16-query tile).  Scores in dynamic LDS.
// ---------------------------------------------------------------------------
__device__ __forceinline__ void ins5(float* t5, float s) {
  if (s > t5[4]) {
    t5[4] = s;
#pragma unroll
    for (int i = 4; i > 0; --i)
      if (t5[i] > t5[i - 1]) { float t = t5[i]; t5[i] = t5[i - 1]; t5[i - 1] = t; }
  }
}

#define LOCAL_SMEM_BYTES (16 * 2048 * 4 + 16 * 16 * 5 * 4 + 16 * 4 * 4 + 16 * 64 * 4 * 2)

__global__ __launch_bounds__(256) void local_attn_kernel(
    const unsigned short* __restrict__ Q,   // [16][2048][64] bf16
    const unsigned short* __restrict__ K,
    const unsigned short* __restrict__ V,
    unsigned short* __restrict__ O) {       // merged [8192][256] bf16
  extern __shared__ char smem_raw[];
  float* sS      = (float*)smem_raw;         // 16*2048
  float* sTop    = sS + 16 * 2048;           // 16*16*5
  float* sThresh = sTop + 16 * 16 * 5;       // 16
  float* sMax    = sThresh + 16;             // 16
  float* sDen    = sMax + 16;                // 16
  int*   sCnt    = (int*)(sDen + 16);        // 16
  int*   sKey    = sCnt + 16;                // 16*64
  float* sWgt    = (float*)(sKey + 16 * 64); // 16*64

  const int bh = blockIdx.x >> 7;
  const int q0 = (blockIdx.x & 127) * 16;
  const int lane = threadIdx.x & 31;
  const int wave = threadIdx.x >> 5;
  const int l16 = lane & 15, h = lane >> 4;
  const size_t base = (size_t)bh * 2048 * 64;

  // Q fragments (A layout): d 0..31 and 32..63
  const unsigned short* qrow = Q + base + (size_t)(q0 + l16) * 64;
  Frag aQ0, aQ1;
  aQ0.q[0] = *(const u32x4*)(qrow + h * 16);
  aQ0.q[1] = *(const u32x4*)(qrow + h * 16 + 8);
  aQ1.q[0] = *(const u32x4*)(qrow + 32 + h * 16);
  aQ1.q[1] = *(const u32x4*)(qrow + 40 + h * 16);

  // Phase 1: scores (each wave covers 256 keys) -> LDS.
  // Two key tiles per iteration: two independent WMMA chains fill the
  // WMMA->WMMA RAW hazard slots.
  for (int kt = 0; kt < 16; kt += 2) {
    const int k0 = wave * 256 + kt * 16;
    const unsigned short* krA = K + base + (size_t)(k0 + l16) * 64 + h * 16;
    const unsigned short* krB = krA + 16 * 64;
    Frag a0, a1, b0, b1;
    a0.q[0] = ((const u32x4*)krA)[0];        a0.q[1] = ((const u32x4*)krA)[1];
    a1.q[0] = ((const u32x4*)(krA + 32))[0]; a1.q[1] = ((const u32x4*)(krA + 32))[1];
    b0.q[0] = ((const u32x4*)krB)[0];        b0.q[1] = ((const u32x4*)krB)[1];
    b1.q[0] = ((const u32x4*)(krB + 32))[0]; b1.q[1] = ((const u32x4*)(krB + 32))[1];
    f32x8 accA = {}, accB = {};
    accA = wmma_bf16(aQ0, a0, accA);
    accB = wmma_bf16(aQ0, b0, accB);
    accA = wmma_bf16(aQ1, a1, accA);
    accB = wmma_bf16(aQ1, b1, accB);
#pragma unroll
    for (int j = 0; j < 8; ++j) {
      sS[(j + h * 8) * 2048 + k0 + l16]      = accA[j] * 0.125f;   // / sqrt(64)
      sS[(j + h * 8) * 2048 + k0 + 16 + l16] = accB[j] * 0.125f;
    }
  }
  __syncthreads();

  // Phase 2: exact top-5 per row (16 threads/row, strided), then merge
  {
    const int r = threadIdx.x >> 4, c = threadIdx.x & 15;
    float t5[5] = {-1e30f, -1e30f, -1e30f, -1e30f, -1e30f};
    for (int k = c; k < 2048; k += 16) ins5(t5, sS[r * 2048 + k]);
#pragma unroll
    for (int i = 0; i < 5; ++i) sTop[(r * 16 + c) * 5 + i] = t5[i];
  }
  __syncthreads();
  if (threadIdx.x < 16) {
    const int r = threadIdx.x;
    float t5[5] = {-1e30f, -1e30f, -1e30f, -1e30f, -1e30f};
    for (int c = 0; c < 16; ++c)
      for (int i = 0; i < 5; ++i) ins5(t5, sTop[(r * 16 + c) * 5 + i]);
    sThresh[r] = t5[4];
    sMax[r]    = t5[0];
    sDen[r]    = 0.f;
    sCnt[r]    = 0;
  }
  __syncthreads();

  // Phase 3a: collect survivors (scores >= thresh), exp weights, denom
  {
    const int r = threadIdx.x >> 4, c = threadIdx.x & 15;
    const float th = sThresh[r], mx = sMax[r];
    for (int k = c; k < 2048; k += 16) {
      const float s = sS[r * 2048 + k];
      if (s >= th) {
        const float w = __expf(s - mx);
        const int idx = atomicAdd(&sCnt[r], 1);
        if (idx < 64) { sKey[r * 64 + idx] = k; sWgt[r * 64 + idx] = w; }
        atomicAdd(&sDen[r], w);
      }
    }
  }
  __syncthreads();

  // Phase 3b: out[r,d] = sum_i w_i * V[k_i, d] / denom  (merged-head layout)
  {
    const int r = threadIdx.x >> 4, c = threadIdx.x & 15;
    int n = sCnt[r]; if (n > 64) n = 64;
    const float inv = 1.0f / sDen[r];
    const int b_ = bh >> 2, hd = bh & 3;
    const size_t obase = ((size_t)(b_ * 2048 + q0 + r)) * 256 + hd * 64;
#pragma unroll
    for (int dd = 0; dd < 4; ++dd) {
      const int d = c * 4 + dd;
      float acc = 0.f;
      for (int i = 0; i < n; ++i)
        acc += sWgt[r * 64 + i] * bf2f(V[base + (size_t)sKey[r * 64 + i] * 64 + d]);
      O[obase + d] = f2bf(acc * inv);
    }
  }
}

// ---------------------------------------------------------------------------
// V transpose: [bh][c][d] -> [bh][d][c]  (enables contiguous V B-fragments)
// ---------------------------------------------------------------------------
__global__ __launch_bounds__(256) void transpose_v_kernel(
    const unsigned short* __restrict__ Vin, unsigned short* __restrict__ Vout) {
  const int idx = blockIdx.x * 256 + threadIdx.x;    // 16*2048*64
  const int d = idx & 63, c = (idx >> 6) & 2047, bh = idx >> 17;
  Vout[((size_t)bh * 64 + d) * 2048 + c] = Vin[idx];
}

// ---------------------------------------------------------------------------
// Global attention: flash-style.  All 8 waves of a block share one bh, so
// K / Vt tiles (32 keys) are staged once per block into LDS with async
// DMA (global_load_async_to_lds_b128, ASYNCcnt), double-buffered so the
// next tile's DMA overlaps this tile's WMMA work.
// ---------------------------------------------------------------------------
__global__ __launch_bounds__(256) void global_attn_kernel(
    const unsigned short* __restrict__ Q,   // [bh][c][d] bf16
    const unsigned short* __restrict__ K,   // [bh][c][d] bf16
    const unsigned short* __restrict__ Vt,  // [bh][d][c] bf16 (transposed)
    unsigned short* __restrict__ O) {       // merged [8192][256] bf16
  __shared__ __align__(16) unsigned short sP[8][16 * 32];     // 8 KB
  __shared__ __align__(16) unsigned short sKbuf[2][32 * 64];  // 2 x 4 KB
  __shared__ __align__(16) unsigned short sVbuf[2][64 * 32];  // 2 x 4 KB
  const int tid  = threadIdx.x;
  const int lane = tid & 31;
  const int wave = tid >> 5;
  const int wid = blockIdx.x * 8 + wave;    // 2048 tiles; same bh per block
  const int bh = wid >> 7;
  const int q0 = (wid & 127) * 16;
  const int l16 = lane & 15, h = lane >> 4;
  const size_t base = (size_t)bh * 2048 * 64;
  const size_t vtbase = (size_t)bh * 64 * 2048;
  unsigned short* myP = &sP[wave][0];

  const unsigned int ldsK0 = lds_off(&sKbuf[0][0]), ldsK1 = lds_off(&sKbuf[1][0]);
  const unsigned int ldsV0 = lds_off(&sVbuf[0][0]), ldsV1 = lds_off(&sVbuf[1][0]);

  // Stage 32-key K tile (contiguous 4 KB) + Vt tile (64 rows x 64 B) via
  // async-to-LDS DMA; one b128 chunk per thread per tile.
  auto stage = [&](int kk, int bufsel) {
    const unsigned int lk = (bufsel ? ldsK1 : ldsK0) + tid * 16;
    const unsigned long long gk =
        (unsigned long long)(size_t)(K + base + (size_t)kk * 64) + (size_t)tid * 16;
    asm volatile("global_load_async_to_lds_b128 %0, %1, off"
                 :: "v"(lk), "v"(gk) : "memory");
    const int d = tid >> 2, ch = tid & 3;
    const unsigned int lv = (bufsel ? ldsV1 : ldsV0) + d * 64 + ch * 16;
    const unsigned long long gv =
        (unsigned long long)(size_t)(Vt + vtbase + (size_t)d * 2048 + kk) + (size_t)ch * 16;
    asm volatile("global_load_async_to_lds_b128 %0, %1, off"
                 :: "v"(lv), "v"(gv) : "memory");
  };

  const unsigned short* qrow = Q + base + (size_t)(q0 + l16) * 64;
  Frag aQ0, aQ1;
  aQ0.q[0] = *(const u32x4*)(qrow + h * 16);
  aQ0.q[1] = *(const u32x4*)(qrow + h * 16 + 8);
  aQ1.q[0] = *(const u32x4*)(qrow + 32 + h * 16);
  aQ1.q[1] = *(const u32x4*)(qrow + 40 + h * 16);

  float rm[8], rl[8];
#pragma unroll
  for (int j = 0; j < 8; ++j) { rm[j] = -1e30f; rl[j] = 0.f; }
  f32x8 accO[4] = {};

  stage(0, 0);
  for (int it = 0; it < 64; ++it) {
    const int k0 = it * 32;
    const int cur = it & 1;
    if (it + 1 < 64) {
      stage(k0 + 32, cur ^ 1);
      // async loads complete in order: <=2 outstanding => current tile landed
      asm volatile("s_wait_asynccnt 0x2" ::: "memory");
    } else {
      asm volatile("s_wait_asynccnt 0x0" ::: "memory");
    }
    __syncthreads();

    const unsigned short* sK = &sKbuf[cur][0];
    const unsigned short* sV = &sVbuf[cur][0];

    f32x8 s0 = {}, s1 = {};
    {
      const unsigned short* kr = sK + l16 * 64 + h * 16;        // keys 0..15
      const unsigned short* kr2 = kr + 16 * 64;                 // keys 16..31
      Frag b0, b1, c0, c1;
      b0.q[0] = ((const u32x4*)kr)[0];         b0.q[1] = ((const u32x4*)kr)[1];
      b1.q[0] = ((const u32x4*)(kr + 32))[0];  b1.q[1] = ((const u32x4*)(kr + 32))[1];
      c0.q[0] = ((const u32x4*)kr2)[0];        c0.q[1] = ((const u32x4*)kr2)[1];
      c1.q[0] = ((const u32x4*)(kr2 + 32))[0]; c1.q[1] = ((const u32x4*)(kr2 + 32))[1];
      s0 = wmma_bf16(aQ0, b0, s0); s1 = wmma_bf16(aQ0, c0, s1);
      s0 = wmma_bf16(aQ1, b1, s0); s1 = wmma_bf16(aQ1, c1, s1);
    }
    float corr[8];
#pragma unroll
    for (int j = 0; j < 8; ++j) {
      const float a = s0[j] * 0.125f, b = s1[j] * 0.125f;
      float tm = fmaxf(a, b);
#pragma unroll
      for (int off = 1; off < 16; off <<= 1) tm = fmaxf(tm, __shfl_xor(tm, off, 32));
      const float nm = fmaxf(rm[j], tm);
      const float p0 = __expf(a - nm), p1 = __expf(b - nm);
      float rs = p0 + p1;
#pragma unroll
      for (int off = 1; off < 16; off <<= 1) rs += __shfl_xor(rs, off, 32);
      corr[j] = __expf(rm[j] - nm);
      rl[j] = rl[j] * corr[j] + rs;
      rm[j] = nm;
      myP[(j + h * 8) * 32 + l16]      = f2bf(p0);   // P row-major 16x32
      myP[(j + h * 8) * 32 + 16 + l16] = f2bf(p1);
    }
#pragma unroll
    for (int t = 0; t < 4; ++t)
#pragma unroll
      for (int j = 0; j < 8; ++j) accO[t][j] *= corr[j];

    __syncthreads();                       // make P visible across lanes
    Frag aP;                               // re-layout P as A fragment
    aP.q[0] = *(const u32x4*)(myP + l16 * 32 + h * 16);
    aP.q[1] = *(const u32x4*)(myP + l16 * 32 + h * 16 + 8);
#pragma unroll
    for (int t = 0; t < 4; ++t) {          // 4 independent WMMA chains
      const unsigned short* vr = sV + (t * 16 + l16) * 32 + h * 16;
      Frag bV;
      bV.q[0] = ((const u32x4*)vr)[0];
      bV.q[1] = ((const u32x4*)vr)[1];
      accO[t] = wmma_bf16(aP, bV, accO[t]);
    }
    __syncthreads();                       // tile fully consumed
  }

  const int b_ = bh >> 2, hd = bh & 3;
#pragma unroll
  for (int t = 0; t < 4; ++t)
#pragma unroll
    for (int j = 0; j < 8; ++j) {
      const int row = j + h * 8;
      const float val = accO[t][j] / rl[j];
      O[((size_t)(b_ * 2048 + q0 + row)) * 256 + hd * 64 + t * 16 + l16] = f2bf(val);
    }
}

// ---------------------------------------------------------------------------
// Residual + LayerNorm (f32): out = LN(x + glob) * g + b
// ---------------------------------------------------------------------------
__global__ __launch_bounds__(256) void layernorm_kernel(
    const float* __restrict__ x, const float* __restrict__ glob,
    const float* __restrict__ g, const float* __restrict__ b,
    float* __restrict__ out) {
  __shared__ float red[8], red2[8];
  const int m = blockIdx.x, t = threadIdx.x;
  const float y = x[(size_t)m * 256 + t] + glob[(size_t)m * 256 + t];
  float s = y, s2 = y * y;
#pragma unroll
  for (int off = 1; off < 32; off <<= 1) {
    s  += __shfl_xor(s, off, 32);
    s2 += __shfl_xor(s2, off, 32);
  }
  if ((t & 31) == 0) { red[t >> 5] = s; red2[t >> 5] = s2; }
  __syncthreads();
  float ts = 0.f, ts2 = 0.f;
#pragma unroll
  for (int i = 0; i < 8; ++i) { ts += red[i]; ts2 += red2[i]; }
  const float mu = ts * (1.0f / 256.0f);
  const float var = ts2 * (1.0f / 256.0f) - mu * mu;
  const float inv = rsqrtf(var + 1e-5f);
  out[(size_t)m * 256 + t] = (y - mu) * inv * g[t] + b[t];
}

// ---------------------------------------------------------------------------
// Launcher
// ---------------------------------------------------------------------------
extern "C" void kernel_launch(void* const* d_in, const int* in_sizes, int n_in,
                              void* d_out, int out_size, void* d_ws, size_t ws_size,
                              hipStream_t stream) {
  (void)in_sizes; (void)n_in; (void)out_size; (void)ws_size;
  const float* x = (const float*)d_in[0];
  const float* Wf[8]; const float* Bf[8];
  for (int i = 0; i < 8; ++i) {               // lq, lk, lv, lo, gq, gk, gv, go
    Wf[i] = (const float*)d_in[1 + 2 * i];
    Bf[i] = (const float*)d_in[2 + 2 * i];
  }
  const float* ln_g = (const float*)d_in[17];
  const float* ln_b = (const float*)d_in[18];

  const int M = 8192;                          // B * C
  char* ws = (char*)d_ws;
  auto carve = [&](size_t bytes) {
    char* p = ws; ws += (bytes + 255) & ~(size_t)255; return p;
  };
  unsigned short* xbf = (unsigned short*)carve((size_t)M * 256 * 2);
  unsigned short* wbf[8];
  for (int i = 0; i < 8; ++i) wbf[i] = (unsigned short*)carve(256 * 256 * 2);
  unsigned short* q1     = (unsigned short*)carve((size_t)M * 256 * 2);
  unsigned short* k1     = (unsigned short*)carve((size_t)M * 256 * 2);
  unsigned short* v1     = (unsigned short*)carve((size_t)M * 256 * 2);
  unsigned short* lmerge = (unsigned short*)carve((size_t)M * 256 * 2);
  unsigned short* lout   = (unsigned short*)carve((size_t)M * 256 * 2);
  unsigned short* q2     = (unsigned short*)carve((size_t)M * 256 * 2);
  unsigned short* k2     = (unsigned short*)carve((size_t)M * 256 * 2);
  unsigned short* v2     = (unsigned short*)carve((size_t)M * 256 * 2);
  unsigned short* v2t    = (unsigned short*)carve((size_t)M * 256 * 2);
  unsigned short* gmerge = (unsigned short*)carve((size_t)M * 256 * 2);
  float*          glob   = (float*)carve((size_t)M * 256 * 4);

  // Precision converts
  f2bf_kernel<<<(M * 256) / 256, 256, 0, stream>>>(x, xbf, M * 256);
  for (int i = 0; i < 8; ++i)
    f2bf_kernel<<<256, 256, 0, stream>>>(Wf[i], wbf[i], 256 * 256);

  // Local-branch projections (head-split layout)
  gemm_bias_kernel<1><<<256, 256, 0, stream>>>(xbf, wbf[0], Bf[0], (void*)q1);
  gemm_bias_kernel<1><<<256, 256, 0, stream>>>(xbf, wbf[1], Bf[1], (void*)k1);
  gemm_bias_kernel<1><<<256, 256, 0, stream>>>(xbf, wbf[2], Bf[2], (void*)v1);

  // Local attention (top-5 masked)
  local_attn_kernel<<<2048, 256, LOCAL_SMEM_BYTES, stream>>>(q1, k1, v1, lmerge);

  // lo projection, then global-branch projections
  gemm_bias_kernel<0><<<256, 256, 0, stream>>>(lmerge, wbf[3], Bf[3], (void*)lout);
  gemm_bias_kernel<1><<<256, 256, 0, stream>>>(lout, wbf[4], Bf[4], (void*)q2);
  gemm_bias_kernel<1><<<256, 256, 0, stream>>>(lout, wbf[5], Bf[5], (void*)k2);
  gemm_bias_kernel<1><<<256, 256, 0, stream>>>(lout, wbf[6], Bf[6], (void*)v2);

  // Global attention (flash, async-LDS double-buffered K/Vt staging)
  transpose_v_kernel<<<(M * 256) / 256, 256, 0, stream>>>(v2, v2t);
  global_attn_kernel<<<256, 256, 0, stream>>>(q2, k2, v2t, gmerge);

  // go projection (f32) + residual LayerNorm
  gemm_bias_kernel<2><<<256, 256, 0, stream>>>(gmerge, wbf[7], Bf[7], (void*)glob);
  layernorm_kernel<<<M, 256, 0, stream>>>(x, glob, ln_g, ln_b, (float*)d_out);
}
